// TransformerBlock_60095182405965
// MI455X (gfx1250) — matrix operand order repaired and checked by reading, not compiled, not run on hardware
//
#include <hip/hip_runtime.h>
#include <hip/hip_bf16.h>
#include <math.h>

// ---------------------------------------------------------------------------
// CDNA5 (gfx1250) wave32 WMMA transformer block.
//   x:[16,192,40,40] f32, heads=6, hd=32, N=1600/batch, P=25600 pixels total.
// All matmuls via v_wmma_f32_16x16x32_f16 (f32 accumulate).
// Attention computed transposed (S^T = K Q^T, O^T = V^T P^T) so softmax
// reductions are per-lane: 2 cross-lane shuffles per 32-key step, no barriers.
// ---------------------------------------------------------------------------

typedef __attribute__((ext_vector_type(16))) _Float16 v16h;
typedef __attribute__((ext_vector_type(8)))  _Float16 v8h;
typedef __attribute__((ext_vector_type(8)))  float    v8f;

#define WMMA16(a, b, c) \
  __builtin_amdgcn_wmma_f32_16x16x32_f16(false, (a), false, (b), (short)0, (c), false, false)

// ---- WMMA fragment loaders (contiguous k order, wave32) -------------------
// A (16x32 f16): lane L -> row m = L&15; slots i: k = 16*(L>>4) + i
// => two 8-half chunks at k = 16*(L>>4) and k = 16*(L>>4) + 8.
static __device__ __forceinline__ v16h frag_a(const _Float16* base, int lda, int lane) {
  const int m  = lane & 15;
  const int hh = (lane >> 4) & 1;
  const _Float16* p = base + (size_t)m * lda + hh * 16;
  union { v16h v; v8h h[2]; } u;
  u.h[0] = *(const v8h*)(p);
  u.h[1] = *(const v8h*)(p + 8);
  return u.v;
}
// B (32x16 f16): lane L -> col n = L&15; slots i: k = 16*(L>>4) + i
// => one contiguous 16-half chunk of row n of W[n,k] (row-major, ldb = K).
static __device__ __forceinline__ v16h frag_b(const _Float16* base, int ldb, int lane) {
  const int n  = lane & 15;
  const int hh = (lane >> 4) & 1;
  return *(const v16h*)(base + (size_t)n * ldb + hh * 16);
}
// C/D (16x16 f32): lane L, reg r -> m = r + 8*(L>>4), n = L&15.

// ---------------------------------------------------------------------------
__global__ void f32_to_f16_kernel(const float* __restrict__ in, _Float16* __restrict__ out, int n) {
  int i = blockIdx.x * 256 + threadIdx.x;
  if (i < n) out[i] = (_Float16)in[i];
}

// LayerNorm over C=192 of x[B,C,1600] (channel-strided) -> pixel-major f16 rows.
__global__ void ln_bcn_kernel(const float* __restrict__ x, const float* __restrict__ g,
                              const float* __restrict__ be, _Float16* __restrict__ xn) {
  int p = blockIdx.x * 256 + threadIdx.x;
  if (p >= 25600) return;
  int b = p / 1600, n = p - b * 1600;
  const float* xb = x + (size_t)b * 192 * 1600 + n;
  float s = 0.f, ss = 0.f;
  for (int c = 0; c < 192; ++c) { float v = xb[(size_t)c * 1600]; s += v; ss += v * v; }
  float mean = s * (1.f / 192.f);
  float var  = ss * (1.f / 192.f) - mean * mean;
  float rstd = rsqrtf(var + 1e-6f);
  _Float16* o = xn + (size_t)p * 192;
  for (int c = 0; c < 192; ++c)
    o[c] = (_Float16)((xb[(size_t)c * 1600] - mean) * rstd * g[c] + be[c]);
}

// LayerNorm over C=192 of pixel-major f32 rows -> pixel-major f16 rows.
__global__ void ln_rows_kernel(const float* __restrict__ x1, const float* __restrict__ g,
                               const float* __restrict__ be, _Float16* __restrict__ xn) {
  int p = blockIdx.x * 256 + threadIdx.x;
  if (p >= 25600) return;
  const float* r = x1 + (size_t)p * 192;
  float s = 0.f, ss = 0.f;
  for (int c = 0; c < 192; ++c) { float v = r[c]; s += v; ss += v * v; }
  float mean = s * (1.f / 192.f);
  float var  = ss * (1.f / 192.f) - mean * mean;
  float rstd = rsqrtf(var + 1e-6f);
  _Float16* o = xn + (size_t)p * 192;
  for (int c = 0; c < 192; ++c)
    o[c] = (_Float16)((r[c] - mean) * rstd * g[c] + be[c]);
}

// ---------------------------------------------------------------------------
// GEMM: out[m,n] = sum_k A[m,k] * W[n,k]  (A pixel-major f16, W row-major f16)
// block = 256 thr (8 waves); block tile 128m x 64n; wave tile 16m x 64n.
enum { GM_QKV = 0, GM_PROJ = 1, GM_GELU = 2, GM_FC2 = 3 };

template <int MODE>
__global__ __launch_bounds__(256) void gemm_kernel(
    const _Float16* __restrict__ A, const _Float16* __restrict__ W,
    const float* __restrict__ bias, const float* __restrict__ aux,
    void* __restrict__ out, void* __restrict__ out2, int N, int K) {
  const int lane = threadIdx.x & 31;
  const int wave = threadIdx.x >> 5;
  const int m0 = blockIdx.x * 128 + wave * 16;
  const int n0 = blockIdx.y * 64;
  const int hh = lane >> 4, nl = lane & 15;

  v8f acc[4];
  const v8f vzero = {0.f, 0.f, 0.f, 0.f, 0.f, 0.f, 0.f, 0.f};
#pragma unroll
  for (int t = 0; t < 4; ++t) acc[t] = vzero;

  for (int k = 0; k < K; k += 32) {
    v16h a = frag_a(A + (size_t)m0 * K + k, K, lane);
#pragma unroll
    for (int t = 0; t < 4; ++t) {
      v16h b = frag_b(W + (size_t)(n0 + 16 * t) * K + k, K, lane);
      acc[t] = WMMA16(a, b, acc[t]);
    }
  }

  const float qscale = 0.17677669529663687f;  // 1/sqrt(32), folded into Q
#pragma unroll
  for (int t = 0; t < 4; ++t) {
    const int c = n0 + 16 * t + nl;
    float bv = (MODE == GM_QKV) ? 0.f : bias[c];
#pragma unroll
    for (int r = 0; r < 8; ++r) {
      const int m = m0 + r + 8 * hh;  // pixel index
      float v = acc[t][r];
      if (MODE == GM_QKV) {
        // q,k -> qk[p,384] (q pre-scaled); v -> vT[b,h,d,j] transposed
        if (c < 384) {
          float qv = (c < 192) ? v * qscale : v;
          ((_Float16*)out)[(size_t)m * 384 + c] = (_Float16)qv;
        } else {
          int b = m / 1600, pn = m - b * 1600;
          int hv = (c - 384) >> 5, dv = (c - 384) & 31;
          ((_Float16*)out2)[((size_t)(b * 6 + hv) * 32 + dv) * 1600 + pn] = (_Float16)v;
        }
      } else if (MODE == GM_PROJ) {
        int b = m / 1600, pn = m - b * 1600;
        float xv = aux[((size_t)b * 192 + c) * 1600 + pn];  // residual from x[B,C,N]
        ((float*)out)[(size_t)m * 192 + c] = v + bv + xv;   // x1 pixel-major f32
      } else if (MODE == GM_GELU) {
        float u = v + bv;
        float gl = 0.5f * u * (1.f + erff(u * 0.70710678118654752f));
        ((_Float16*)out)[(size_t)m * N + c] = (_Float16)gl;
      } else {  // GM_FC2: +bias + x1 residual, transpose to [B,C,N]
        int b = m / 1600, pn = m - b * 1600;
        float res = v + bv + aux[(size_t)m * 192 + c];
        ((float*)out)[((size_t)b * 192 + c) * 1600 + pn] = res;
      }
    }
  }
}

// ---------------------------------------------------------------------------
// Flash attention (transposed): qk [p,384] f16 (q|k rows), vT [b,h,32d,1600j].
// block = 128 thr (4 waves); block = (b, h, 64 queries); wave = 16 queries.
// S^T = K Q^T  -> lane holds (j = r+8hh, q = lane&15): softmax is per-lane.
// O^T = V^T P^T -> lane holds (d = r+8hh, q = lane&15): rescale is per-lane.
__global__ __launch_bounds__(128) void attn_kernel(const _Float16* __restrict__ qk,
                                                   const _Float16* __restrict__ vT,
                                                   _Float16* __restrict__ o_t) {
  __shared__ __align__(32) _Float16 Pl[4][16 * 32];  // per-wave P: [q][j]

  const int lane = threadIdx.x & 31;
  const int wave = threadIdx.x >> 5;
  const int hh = lane >> 4, nl = lane & 15;
  const int b = blockIdx.z, h = blockIdx.y;
  const int q0 = blockIdx.x * 64 + wave * 16;

  const v8f vzero = {0.f, 0.f, 0.f, 0.f, 0.f, 0.f, 0.f, 0.f};

  // Q^T B-fragment: Q rows [q, d] row-major are exactly the WMMA-B layout.
  const v16h bq = frag_b(qk + (size_t)(b * 1600 + q0) * 384 + h * 32, 384, lane);
  const _Float16* Kb = qk + (size_t)(b * 1600) * 384 + 192 + h * 32;
  const _Float16* Vb = vT + (size_t)(b * 6 + h) * 32 * 1600;
  _Float16* Pw = &Pl[wave][0];

  v8f o0 = vzero, o1 = vzero;        // O^T tiles: d = r+8hh (+16), q = nl
  float mrow = -1e30f, lrow = 0.f;   // softmax state for q = q0 + nl

  for (int j0 = 0; j0 < 1600; j0 += 32) {
    // S^T tiles: A = K rows (direct global), B = Q^T (preloaded)
    v16h ak0 = frag_a(Kb + (size_t)j0 * 384, 384, lane);
    v16h ak1 = frag_a(Kb + (size_t)(j0 + 16) * 384, 384, lane);
    v8f s0 = WMMA16(ak0, bq, vzero);  // (j = j0    + r+8hh, q = nl)
    v8f s1 = WMMA16(ak1, bq, vzero);  // (j = j0+16 + r+8hh, q = nl)

    // per-lane max over this lane's 16 j's, then one xor-16 to cover all 32
    float mx = fmaxf(s0[0], s1[0]);
#pragma unroll
    for (int r = 1; r < 8; ++r) mx = fmaxf(mx, fmaxf(s0[r], s1[r]));
    mx = fmaxf(mx, __shfl_xor(mx, 16));
    float mnew = fmaxf(mrow, mx);
    float corr = __expf(mrow - mnew);

    float p0[8], p1[8], rs = 0.f;
#pragma unroll
    for (int r = 0; r < 8; ++r) {
      p0[r] = __expf(s0[r] - mnew);
      p1[r] = __expf(s1[r] - mnew);
      rs += p0[r] + p1[r];
    }
    rs += __shfl_xor(rs, 16);
    lrow = lrow * corr + rs;
    mrow = mnew;
#pragma unroll
    for (int r = 0; r < 8; ++r) { o0[r] *= corr; o1[r] *= corr; }

    // stage P^T -> Pl[q][j]: two contiguous b128 stores (same-wave LDS, in order)
    v8h w0, w1;
#pragma unroll
    for (int r = 0; r < 8; ++r) { w0[r] = (_Float16)p0[r]; w1[r] = (_Float16)p1[r]; }
    *(v8h*)(Pw + nl * 32 + 8 * hh)      = w0;
    *(v8h*)(Pw + nl * 32 + 16 + 8 * hh) = w1;

    v16h bp  = frag_b(Pw, 32, lane);                            // P^T as B
    v16h av0 = frag_a(Vb + j0, 1600, lane);                     // V^T rows d=0..15
    v16h av1 = frag_a(Vb + (size_t)16 * 1600 + j0, 1600, lane); // d=16..31
    o0 = WMMA16(av0, bp, o0);
    o1 = WMMA16(av1, bp, o1);
  }

  // O^T -> o_t[p,192]: per lane q = q0+nl fixed, d = r+8hh contiguous
  float inv = 1.f / lrow;
  _Float16* op = o_t + (size_t)(b * 1600 + q0 + nl) * 192 + h * 32;
  v8h r0, r1;
#pragma unroll
  for (int r = 0; r < 8; ++r) {
    r0[r] = (_Float16)(o0[r] * inv);
    r1[r] = (_Float16)(o1[r] * inv);
  }
  *(v8h*)(op + 8 * hh)      = r0;
  *(v8h*)(op + 16 + 8 * hh) = r1;
}

// ---------------------------------------------------------------------------
extern "C" void kernel_launch(void* const* d_in, const int* in_sizes, int n_in,
                              void* d_out, int out_size, void* d_ws, size_t ws_size,
                              hipStream_t stream) {
  const float* x      = (const float*)d_in[0];
  const float* w_qkv  = (const float*)d_in[1];
  const float* w_proj = (const float*)d_in[2];
  const float* b_proj = (const float*)d_in[3];
  const float* w_fc1  = (const float*)d_in[4];
  const float* b_fc1  = (const float*)d_in[5];
  const float* w_fc2  = (const float*)d_in[6];
  const float* b_fc2  = (const float*)d_in[7];
  const float* g1     = (const float*)d_in[8];
  const float* beta1  = (const float*)d_in[9];
  const float* g2     = (const float*)d_in[10];
  const float* beta2  = (const float*)d_in[11];

  const int P = 25600;  // 16 * 1600 pixels

  char* ws = (char*)d_ws;
  size_t off = 0;
  auto carve = [&](size_t bytes) -> char* {
    char* p = ws + off;
    off = (off + bytes + 255) & ~(size_t)255;
    return p;
  };
  _Float16* wq16 = (_Float16*)carve((size_t)576 * 192 * 2);
  _Float16* wp16 = (_Float16*)carve((size_t)192 * 192 * 2);
  _Float16* w116 = (_Float16*)carve((size_t)768 * 192 * 2);
  _Float16* w216 = (_Float16*)carve((size_t)192 * 768 * 2);
  _Float16* xn1  = (_Float16*)carve((size_t)P * 192 * 2);
  _Float16* qk   = (_Float16*)carve((size_t)P * 384 * 2);      // q|k, pixel rows
  _Float16* vT   = (_Float16*)carve((size_t)16 * 6 * 32 * 1600 * 2);  // [b,h,d,j]
  _Float16* ot   = (_Float16*)carve((size_t)P * 192 * 2);
  float*    x1   = (float*)   carve((size_t)P * 192 * 4);
  _Float16* xn2  = (_Float16*)carve((size_t)P * 192 * 2);
  _Float16* h1   = (_Float16*)carve((size_t)P * 768 * 2);
  (void)ws_size; (void)in_sizes; (void)n_in; (void)out_size;

  // 1) weights -> f16
  f32_to_f16_kernel<<<(576 * 192 + 255) / 256, 256, 0, stream>>>(w_qkv, wq16, 576 * 192);
  f32_to_f16_kernel<<<(192 * 192 + 255) / 256, 256, 0, stream>>>(w_proj, wp16, 192 * 192);
  f32_to_f16_kernel<<<(768 * 192 + 255) / 256, 256, 0, stream>>>(w_fc1, w116, 768 * 192);
  f32_to_f16_kernel<<<(192 * 768 + 255) / 256, 256, 0, stream>>>(w_fc2, w216, 192 * 768);

  // 2) LN1: x[B,C,N] -> xn1 [p,192] f16
  ln_bcn_kernel<<<P / 256, 256, 0, stream>>>(x, g1, beta1, xn1);

  // 3) qkv GEMM: q(scaled)|k -> qk [p,384]; v -> vT [b,h,d,j]
  gemm_kernel<GM_QKV><<<dim3(P / 128, 576 / 64), 256, 0, stream>>>(
      xn1, wq16, nullptr, nullptr, qk, vT, 576, 192);

  // 4) flash attention -> ot [p,192] f16
  attn_kernel<<<dim3(1600 / 64, 6, 16), 128, 0, stream>>>(qk, vT, ot);

  // 5) proj + bias + residual(x) -> x1 [p,192] f32
  gemm_kernel<GM_PROJ><<<dim3(P / 128, 192 / 64), 256, 0, stream>>>(
      ot, wp16, b_proj, x, x1, nullptr, 192, 192);

  // 6) LN2 on x1 rows -> xn2 f16
  ln_rows_kernel<<<P / 256, 256, 0, stream>>>(x1, g2, beta2, xn2);

  // 7) fc1 + bias + exact GELU -> h1 [p,768] f16
  gemm_kernel<GM_GELU><<<dim3(P / 128, 768 / 64), 256, 0, stream>>>(
      xn2, w116, b_fc1, nullptr, h1, nullptr, 768, 192);

  // 8) fc2 + bias + residual(x1), transpose to [B,C,H,W] f32 output
  gemm_kernel<GM_FC2><<<dim3(P / 128, 192 / 64), 256, 0, stream>>>(
      h1, w216, b_fc2, x1, d_out, nullptr, 192, 768);
}
